// MonarchAttention_24558622998775
// MI455X (gfx1250) — matrix operand order repaired and checked by reading, not compiled, not run on hardware
//
#include <hip/hip_runtime.h>

typedef __bf16 bf16_t;
typedef __bf16 v16bf __attribute__((ext_vector_type(16)));
typedef float  v8f   __attribute__((ext_vector_type(8)));
typedef unsigned int u32x4 __attribute__((ext_vector_type(4)));
typedef float  f32x4 __attribute__((ext_vector_type(4)));

union Frag16 { v16bf v; u32x4 q[2]; };
union Pack8  { u32x4 q; bf16_t h[8]; };
union Pack2  { unsigned int u; bf16_t h[2]; };

#define RS 72                         // padded LDS row stride (elems): 16B aligned, bank-spread
#define TILE_ELEMS (64 * 16 * RS)     // 73728 elems
#define TILE_BYTES (TILE_ELEMS * 2)   // 147456 B
#define YS_STRIDE 1028                // f32 output staging stride (bank-spread)
#define WSZ 262144                    // 64^3 elems per weight tensor

// ---------------- weight fp32 -> bf16 prologue ----------------
__global__ void __launch_bounds__(256) cvt_weights_kernel(
    const float* __restrict__ a, const float* __restrict__ b,
    const float* __restrict__ c, const float* __restrict__ d,
    bf16_t* __restrict__ o) {
  int i = blockIdx.x * 256 + threadIdx.x;   // 1024 blocks * 256 = 262144
  o[i]            = (bf16_t)a[i];
  o[i + WSZ]      = (bf16_t)b[i];
  o[i + 2 * WSZ]  = (bf16_t)c[i];
  o[i + 3 * WSZ]  = (bf16_t)d[i];
}

// Stage L of a monarch: T[n][ch][b] = sum_m L[b][n][m] * Xsrc[b][ch][m]
// A = weight rows (M=n, K=m) streamed from global; B = data (K=m, N=ch) from LDS.
__device__ __forceinline__ void stageL(const bf16_t* __restrict__ Xsrc,
                                       bf16_t* __restrict__ Tdst,
                                       const bf16_t* __restrict__ wL,
                                       int w, int l15, int hi) {
  for (int bi = 0; bi < 8; ++bi) {
    int b = w + bi * 8;
    const bf16_t* xrow = Xsrc + (b * 16 + l15) * RS + hi * 16;
    const bf16_t* wrow = wL + (b * 64 + l15) * 64 + hi * 16;
    bf16_t* trow = Tdst + (hi * 8 * 16 + l15) * RS + b;
    Frag16 fb[2];
#pragma unroll
    for (int k = 0; k < 2; ++k) {
      fb[k].q[0] = *(const u32x4*)(xrow + k * 32);
      fb[k].q[1] = *(const u32x4*)(xrow + k * 32 + 8);
    }
#pragma unroll
    for (int nt = 0; nt < 4; ++nt) {
      v8f acc = {};
#pragma unroll
      for (int k = 0; k < 2; ++k) {
        Frag16 fa;
        const bf16_t* p = wrow + nt * (16 * 64) + k * 32;
        fa.q[0] = *(const u32x4*)p;
        fa.q[1] = *(const u32x4*)(p + 8);
        acc = __builtin_amdgcn_wmma_f32_16x16x32_bf16(false, fa.v, false, fb[k].v,
                                                      (short)0, acc, false, false);
      }
      // D: M=n over vgprs, N=ch over lanes -> scatter with constant DS offsets
      bf16_t* tp = trow + nt * (16 * 16 * RS);
#pragma unroll
      for (int v2 = 0; v2 < 8; ++v2) {
        tp[v2 * (16 * RS)] = (bf16_t)acc[v2];
      }
    }
  }
}

__global__ void __launch_bounds__(256) monarch_attn_kernel(
    const float* __restrict__ q, const float* __restrict__ kk,
    const float* __restrict__ vvp, const float* __restrict__ b1,
    const float* __restrict__ b2, const bf16_t* __restrict__ wts,
    float* __restrict__ out) {
  extern __shared__ __align__(16) char smem[];
  bf16_t* Xs = (bf16_t*)smem;                 // X tile / X2 tile / (aliased) Ys
  bf16_t* Ts = (bf16_t*)(smem + TILE_BYTES);  // T tile / T2 tile
  float*  Ys = (float*)smem;

  const int t = threadIdx.x;
  const int lane = t & 31, l15 = lane & 15, hi = lane >> 4;
  const int w = t >> 5;

  const int bid   = blockIdx.x;       // 256 blocks = 8 batch * 8 head * 4 dtiles
  const int batch = bid >> 5;
  const int head  = (bid >> 2) & 7;
  const int d0    = (bid & 3) * 16;

  const bf16_t* wL1 = wts;
  const bf16_t* wR1 = wts + WSZ;
  const bf16_t* wL2 = wts + 2 * WSZ;
  const bf16_t* wR2 = wts + 3 * WSZ;

  // flat index for (s, ch):  chanBase + s*512 + ch    (layout B,S,H,D)
  const int chanBase = batch * (4096 * 512) + head * 64 + d0;
  const long outBase = ((long)(batch * 8 + head) * 64 + d0) * 4096;

  // ---- Phase 0: load q slice -> Xs[j][ch][m] (bf16), packed-pair stores ----
  {
    int ch4  = (t & 3) * 4;
    int sIdx = t >> 2;                 // 0..63  (== j)
    const float* qb = q + chanBase + sIdx * 512 + ch4;
    bf16_t* xb = Xs + (sIdx * 16 + ch4) * RS;
    for (int it = 0; it < 32; ++it) {
      // s0 = it*128 + sIdx, s1 = s0 + 64: same j=sIdx, m = 2*it, 2*it+1
      f32x4 f0 = *(const f32x4*)(qb + it * (128 * 512));
      f32x4 f1 = *(const f32x4*)(qb + it * (128 * 512) + 64 * 512);
      Pack2 p0, p1, p2, p3;
      p0.h[0] = (bf16_t)f0.x; p0.h[1] = (bf16_t)f1.x;
      p1.h[0] = (bf16_t)f0.y; p1.h[1] = (bf16_t)f1.y;
      p2.h[0] = (bf16_t)f0.z; p2.h[1] = (bf16_t)f1.z;
      p3.h[0] = (bf16_t)f0.w; p3.h[1] = (bf16_t)f1.w;
      *(unsigned int*)(xb + it * 2)           = p0.u;
      *(unsigned int*)(xb + RS + it * 2)      = p1.u;
      *(unsigned int*)(xb + 2 * RS + it * 2)  = p2.u;
      *(unsigned int*)(xb + 3 * RS + it * 2)  = p3.u;
    }
  }
  __syncthreads();

  // ---- Phase 1: monarch1 stage L:  Xs -> Ts ----
  stageL(Xs, Ts, wL1, w, l15, hi);
  __syncthreads();

  // ---- Phase 2: monarch1 stage R + (*k + b1) -> X2s (aliases Xs) ----
  for (int ni = 0; ni < 8; ++ni) {
    int n = w + ni * 8;
    const bf16_t* trow = Ts + (n * 16 + l15) * RS + hi * 16;
    const bf16_t* wrow = wR1 + (n * 64 + l15) * 64 + hi * 16;
    Frag16 fb[2];
#pragma unroll
    for (int k = 0; k < 2; ++k) {
      fb[k].q[0] = *(const u32x4*)(trow + k * 32);
      fb[k].q[1] = *(const u32x4*)(trow + k * 32 + 8);
    }
#pragma unroll
    for (int ct = 0; ct < 4; ++ct) {
      v8f acc = {};
#pragma unroll
      for (int k = 0; k < 2; ++k) {
        Frag16 fa;
        const bf16_t* p = wrow + ct * (16 * 64) + k * 32;
        fa.q[0] = *(const u32x4*)p;
        fa.q[1] = *(const u32x4*)(p + 8);
        acc = __builtin_amdgcn_wmma_f32_16x16x32_bf16(false, fa.v, false, fb[k].v,
                                                      (short)0, acc, false, false);
      }
      // D: M=c over vgprs (contiguous dst), N=ch over lanes
      Pack8 pk;
      int ch = l15;
      int sbase = (ct * 16 + hi * 8) * 64 + n;
      const float* kp = kk + chanBase + sbase * 512 + ch;
#pragma unroll
      for (int v2 = 0; v2 < 8; ++v2) {
        float kval = kp[v2 * (64 * 512)];
        pk.h[v2] = (bf16_t)(acc[v2] * kval + b1[sbase + v2 * 64]);
      }
      *(u32x4*)(Xs + (n * 16 + ch) * RS + ct * 16 + hi * 8) = pk.q;
    }
  }
  __syncthreads();

  // ---- Phase 3: monarch2 stage L:  X2s -> Ts2 (aliases Ts) ----
  stageL(Xs, Ts, wL2, w, l15, hi);
  __syncthreads();

  // ---- Phase 4: monarch2 stage R + b2, *v, coalesced store via LDS ----
  for (int ct = 0; ct < 4; ++ct) {
    for (int ni = 0; ni < 8; ++ni) {
      int n = w + ni * 8;
      const bf16_t* trow = Ts + (n * 16 + l15) * RS + hi * 16;
      const bf16_t* wrow = wR2 + (n * 64 + l15) * 64 + hi * 16;
      Frag16 fb[2];
#pragma unroll
      for (int k = 0; k < 2; ++k) {
        fb[k].q[0] = *(const u32x4*)(trow + k * 32);
        fb[k].q[1] = *(const u32x4*)(trow + k * 32 + 8);
      }
      v8f acc = {};
#pragma unroll
      for (int k = 0; k < 2; ++k) {
        Frag16 fa;
        const bf16_t* p = wrow + ct * (16 * 64) + k * 32;
        fa.q[0] = *(const u32x4*)p;
        fa.q[1] = *(const u32x4*)(p + 8);
        acc = __builtin_amdgcn_wmma_f32_16x16x32_bf16(false, fa.v, false, fb[k].v,
                                                      (short)0, acc, false, false);
      }
      int ch = l15;
      int sbase = (ct * 16 + hi * 8) * 64 + n;
      const float* vp = vvp + chanBase + sbase * 512 + ch;
      float* yp = Ys + ch * YS_STRIDE + hi * (8 * 64) + n;   // + v2*64 per acc reg
#pragma unroll
      for (int v2 = 0; v2 < 8; ++v2) {
        float vval = vp[v2 * (64 * 512)];
        yp[v2 * 64] = (acc[v2] + b2[sbase + v2 * 64]) * vval;
      }
    }
    __syncthreads();
    // flush 16 x 4KB contiguous chunks:  out[d0+ch][ct*1024 + e]
#pragma unroll
    for (int cc = 0; cc < 2; ++cc) {
      int ch = w * 2 + cc;
      const float* src = Ys + ch * YS_STRIDE;
      float* dst = out + outBase + (long)ch * 4096 + ct * 1024;
#pragma unroll
      for (int i = 0; i < 8; ++i) {
        int e = i * 128 + lane * 4;
        *(f32x4*)(dst + e) = *(const f32x4*)(src + e);
      }
    }
    __syncthreads();
  }
}

extern "C" void kernel_launch(void* const* d_in, const int* in_sizes, int n_in,
                              void* d_out, int out_size, void* d_ws, size_t ws_size,
                              hipStream_t stream) {
  (void)in_sizes; (void)n_in; (void)out_size; (void)ws_size;
  const float* q   = (const float*)d_in[0];
  const float* k   = (const float*)d_in[1];
  const float* v   = (const float*)d_in[2];
  const float* M1L = (const float*)d_in[3];
  const float* M1R = (const float*)d_in[4];
  const float* M2L = (const float*)d_in[5];
  const float* M2R = (const float*)d_in[6];
  const float* b1  = (const float*)d_in[7];
  const float* b2  = (const float*)d_in[8];
  bf16_t* wts = (bf16_t*)d_ws;   // 4 * 512KB bf16 weights

  cvt_weights_kernel<<<1024, 256, 0, stream>>>(M1L, M1R, M2L, M2R, wts);
  monarch_attn_kernel<<<256, 256, 2 * TILE_BYTES, stream>>>(
      q, k, v, b1, b2, wts, (float*)d_out);
}
